// Lstm_66838281061299
// MI455X (gfx1250) — matrix operand order repaired and checked by reading, not compiled, not run on hardware
//
#include <hip/hip_runtime.h>
#include <hip/hip_bf16.h>

typedef _Float16 v16h __attribute__((ext_vector_type(16)));
typedef _Float16 v8h  __attribute__((ext_vector_type(8)));
typedef float    v8f  __attribute__((ext_vector_type(8)));

#define T_STEPS 4096
#define BATCH   64
#define HDIM    60
#define GDIM    240   // 4*H
#define HPAD    64    // K padded to 2 x 32 for wmma f16
#define H32P    65    // pad so column reads of h32 are bank-conflict-free
#define NTH     480   // 15 waves (wave32): one wave per 16-wide N tile

__device__ __forceinline__ float fast_rcp(float z) {
#if __has_builtin(__builtin_amdgcn_rcpf)
    return __builtin_amdgcn_rcpf(z);     // v_rcp_f32 (1 trans op, no IEEE div expansion)
#else
    return 1.0f / z;
#endif
}
__device__ __forceinline__ float sigf(float z) {
    return fast_rcp(1.0f + __expf(-z));  // v_exp_f32 + add + v_rcp_f32
}
__device__ __forceinline__ float tanhfast(float z) {
#if __has_builtin(__builtin_amdgcn_tanhf)
    return __builtin_amdgcn_tanhf(z);    // native V_TANH_F32 on gfx1250
#else
    float e = __expf(2.0f * z);
    return (e - 1.0f) * fast_rcp(e + 1.0f);
#endif
}

__global__ __launch_bounds__(NTH) void lstm_persistent_kernel(
    const float* __restrict__ x,    // [T,B,1]
    const float* __restrict__ h0,   // [1,B,H]
    const float* __restrict__ c0,   // [1,B,H]
    const float* __restrict__ W_ih, // [4H,1]
    const float* __restrict__ W_hh, // [4H,H]
    const float* __restrict__ b_ih, // [4H]
    const float* __restrict__ b_hh, // [4H]
    const float* __restrict__ Wo,   // [1,H]
    const float* __restrict__ bo,   // [1]
    float* __restrict__ out)        // [T,B,1]
{
    // Static LDS: 8K + 16.25K + 30K + ~2.4K  ≈ 57 KB (well under WGP LDS)
    __shared__ __align__(16) _Float16 h16[BATCH * HPAD];   // h state, f16, K-padded (cols 60..63 stay 0)
    __shared__ float    h32[BATCH * H32P];                 // h state, f32, for output projection
    __shared__ _Float16 gates[BATCH * GDIM];               // pre-activation h@W_hh^T
    __shared__ float    WihL[GDIM], bsumL[GDIM], WoL[HPAD], xL[BATCH];

    const int tid  = threadIdx.x;
    const int lane = tid & 31;
    const int w    = tid >> 5;        // wave id == N-tile id (0..14)
    const int g0   = w * 16;          // first gate column of this wave's tile

    // ---------------- one-time init ----------------
    for (int idx = tid; idx < BATCH * HPAD; idx += NTH) {
        int b = idx >> 6, k = idx & 63;
        h16[idx] = (k < HDIM) ? (_Float16)h0[b * HDIM + k] : (_Float16)0.0f;
    }
    if (tid < GDIM) { WihL[tid] = W_ih[tid]; bsumL[tid] = b_ih[tid] + b_hh[tid]; }
    if (tid < HPAD) WoL[tid] = (tid < HDIM) ? Wo[tid] : 0.0f;

    // cell state lives in registers for the whole sequence
    float creg[8];
    #pragma unroll
    for (int j = 0; j < 8; ++j) {
        int e = tid + j * NTH; int b = e & 63; int hh = e >> 6;
        creg[j] = c0[b * HDIM + hh];
    }

    // Preload B fragments of W_hh^T (one N-tile per wave), f32 -> f16, K padded.
    // 16-bit B 32x16 layout: lane L holds N=(L&15), K = kb..kb+7 and kb+16..kb+23 (kb = 0/8 per half)
    v16h bfrag[2];
    #pragma unroll
    for (int kk = 0; kk < 2; ++kk) {
        int kb = kk * 32 + ((lane >> 4) << 3);
        int g  = g0 + (lane & 15);
        #pragma unroll
        for (int i = 0; i < 16; ++i) {
            int k = kb + i + ((i >= 8) ? 8 : 0);
            bfrag[kk][i] = (k < HDIM) ? (_Float16)W_hh[g * HDIM + k] : (_Float16)0.0f;
        }
    }

    __syncthreads();

    const int   rowA    = lane & 15;
    const int   halfOff = (lane >> 4) << 3;   // A-operand K offset: 0 or 8
    const float bias_out = bo[0];

    for (int t = 0; t < T_STEPS; ++t) {
        // stage x(t) into LDS; warm future x in near caches (consumed 8 steps later)
        if (tid < BATCH) xL[tid] = x[t * BATCH + tid];
        if (tid == 0 && t + 8 < T_STEPS) __builtin_prefetch(x + (t + 8) * BATCH, 0, 3);

        // ---- Phase A: gates_pre = h @ W_hh^T  (8 x v_wmma_f32_16x16x32_f16 per wave) ----
        #pragma unroll
        for (int m = 0; m < 4; ++m) {
            v8f acc = {0.f, 0.f, 0.f, 0.f, 0.f, 0.f, 0.f, 0.f};
            #pragma unroll
            for (int kk = 0; kk < 2; ++kk) {
                int row   = m * 16 + rowA;
                int kbase = kk * 32 + halfOff;
                v8h lo = *(const v8h*)&h16[row * HPAD + kbase];
                v8h hi = *(const v8h*)&h16[row * HPAD + kbase + 16];
                v16h a;
                #pragma unroll
                for (int i = 0; i < 8; ++i) { a[i] = lo[i]; a[i + 8] = hi[i]; }
                acc = __builtin_amdgcn_wmma_f32_16x16x32_f16(
                          false, a, false, bfrag[kk], (short)0, acc, false, false);
            }
            // D tile -> LDS (f16). VGPR r: lanes 0-15 row m*16+r, lanes 16-31 row m*16+8+r
            int colD  = g0 + (lane & 15);
            int rbase = m * 16 + ((lane >> 4) << 3);
            #pragma unroll
            for (int r = 0; r < 8; ++r)
                gates[(rbase + r) * GDIM + colD] = (_Float16)acc[r];
        }
        __syncthreads();

        // ---- Phase B: elementwise LSTM cell update (c in registers) ----
        #pragma unroll
        for (int j = 0; j < 8; ++j) {
            int e = tid + j * NTH; int b = e & 63; int hh = e >> 6;
            float xv = xL[b];
            float zi = (float)gates[b * GDIM + hh]       + xv * WihL[hh]       + bsumL[hh];
            float zf = (float)gates[b * GDIM + 60 + hh]  + xv * WihL[60 + hh]  + bsumL[60 + hh];
            float zg = (float)gates[b * GDIM + 120 + hh] + xv * WihL[120 + hh] + bsumL[120 + hh];
            float zo = (float)gates[b * GDIM + 180 + hh] + xv * WihL[180 + hh] + bsumL[180 + hh];
            float ig = sigf(zi), fg = sigf(zf), og = sigf(zo);
            float gg = tanhfast(zg);
            float c  = fg * creg[j] + ig * gg;
            creg[j]  = c;
            float hv = og * tanhfast(c);
            h32[b * H32P + hh] = hv;
            h16[b * HPAD + hh] = (_Float16)hv;   // pad cols 60..63 remain zero
        }
        __syncthreads();

        // ---- Phase C: out[t,b] = h . Wo + bo  (2 waves; overlaps next Phase A) ----
        if (tid < BATCH) {
            float acc = bias_out;
            #pragma unroll 4
            for (int k = 0; k < HDIM; ++k)
                acc += h32[tid * H32P + k] * WoL[k];
            out[t * BATCH + tid] = acc;
        }
    }
}

extern "C" void kernel_launch(void* const* d_in, const int* in_sizes, int n_in,
                              void* d_out, int out_size, void* d_ws, size_t ws_size,
                              hipStream_t stream) {
    const float* x    = (const float*)d_in[0];
    const float* h0   = (const float*)d_in[1];
    const float* c0   = (const float*)d_in[2];
    const float* W_ih = (const float*)d_in[3];
    const float* W_hh = (const float*)d_in[4];
    const float* b_ih = (const float*)d_in[5];
    const float* b_hh = (const float*)d_in[6];
    const float* Wo   = (const float*)d_in[7];
    const float* bo   = (const float*)d_in[8];
    float* out = (float*)d_out;

    // Single persistent workgroup: the recurrence is strictly serial, so all
    // state is pinned to one WGP; 15 wave32s cover the 15 N-tiles of 4H=240.
    lstm_persistent_kernel<<<1, NTH, 0, stream>>>(x, h0, c0, W_ih, W_hh,
                                                  b_ih, b_hh, Wo, bo, out);
}
